// MotionNet_29016799052669
// MI455X (gfx1250) — matrix operand order repaired and checked by reading, not compiled, not run on hardware
//
#include <hip/hip_runtime.h>

// ---------------------------------------------------------------------------
// MI455X / gfx1250: implicit-GEMM 3x3 convs on v_wmma_f32_16x16x32_bf16.
// NHWC activations with a 1-pixel ZERO HALO (258x258) so conv A-fragment
// loads are branch-free, and channel counts padded to multiples of 32 so each
// WMMA K-chunk is one contiguous slab. Weights staged in LDS per block
// (ds_load_b128 B-fragments). Correlation + bilinear upsample fused into the
// NHWC packing prologue.
// ---------------------------------------------------------------------------

typedef __bf16 bf16_t;
typedef __attribute__((ext_vector_type(16))) __bf16 v16bf;
typedef __attribute__((ext_vector_type(8)))  __bf16 v8bf;
typedef __attribute__((ext_vector_type(8)))  float  v8f;

#define HH 256
#define WW 256
#define BB 4
#define HW (HH * WW)
#define PH 258                    // padded height
#define PW 258                    // padded width
#define PPX (PH * PW)             // padded pixels per batch

// f32 -> bf16 round-to-nearest-even
__device__ __forceinline__ unsigned short f2bf(float f) {
    unsigned int u = __float_as_uint(f);
    u += 0x7FFFu + ((u >> 16) & 1u);
    return (unsigned short)(u >> 16);
}
__device__ __forceinline__ float bf2f(unsigned short s) {
    return __uint_as_float(((unsigned int)s) << 16);
}
__device__ __forceinline__ float lrelu(float v) {
    return v >= 0.0f ? v : 0.1f * v;
}

// ---------------------------------------------------------------------------
// Weight repack: [O][I][3][3] f32  ->  [tap(9)][O][Ipad] bf16
// ---------------------------------------------------------------------------
__global__ void prep_weights_k(const float* __restrict__ w0,
                               const float* __restrict__ w1,
                               const float* __restrict__ w2,
                               const float* __restrict__ wf,
                               unsigned short* __restrict__ w0r,   // [9][32][128]
                               unsigned short* __restrict__ w1r,   // [9][32][32]
                               unsigned short* __restrict__ w2r,   // [9][32][32]
                               unsigned short* __restrict__ wfr) { // [9][16][32]
    int tid = blockIdx.x * blockDim.x + threadIdx.x;
    int nthreads = gridDim.x * blockDim.x;
    for (int i = tid; i < 9 * 32 * 128; i += nthreads) {
        int c = i & 127, o = (i >> 7) & 31, t = i >> 12;
        float v = (c < 115) ? w0[(o * 115 + c) * 9 + t] : 0.0f;
        w0r[i] = f2bf(v);
    }
    for (int i = tid; i < 9 * 32 * 32; i += nthreads) {
        int c = i & 31, o = (i >> 5) & 31, t = i >> 10;
        w1r[i] = f2bf(w1[(o * 32 + c) * 9 + t]);
        w2r[i] = f2bf(w2[(o * 32 + c) * 9 + t]);
    }
    for (int i = tid; i < 9 * 16 * 32; i += nthreads) {
        int c = i & 31, o = (i >> 5) & 15, t = i >> 9;
        float v = (o < 3) ? wf[(o * 32 + c) * 9 + t] : 0.0f;
        wfr[i] = f2bf(v);
    }
}

// ---------------------------------------------------------------------------
// Zero the 1-pixel halo of a padded NHWC bf16 tensor [B][PH][PW][cin].
// Interior is fully overwritten by producers, so only the border needs it.
// ---------------------------------------------------------------------------
__global__ void zero_halo_k(unsigned short* __restrict__ t, int cin) {
    int idx = blockIdx.x * blockDim.x + threadIdx.x;   // over B*PPX
    if (idx >= BB * PPX) return;
    int p = idx % PPX;
    int y = p / PW, x = p % PW;
    if (y == 0 || y == PH - 1 || x == 0 || x == PW - 1) {
        uint4* d = (uint4*)(t + (size_t)idx * cin);
        uint4 z = {0u, 0u, 0u, 0u};
        int n = cin >> 3;                              // 8 shorts per uint4
        for (int c = 0; c < n; ++c) d[c] = z;
    }
}

// ---------------------------------------------------------------------------
// Fused correlation + lrelu + bilinear upsample(feature, flow*2) -> padded
// NHWC bf16. inp[b][y+1][x+1][c]: 0..80 corr, 81..112 feat, 113..114 flow,
// 115..127 zero.
// ---------------------------------------------------------------------------
__global__ void build_input_k(const float* __restrict__ xr,
                              const float* __restrict__ xn,
                              const float* __restrict__ bflow,  // [B,1,2,128,128]
                              const float* __restrict__ bfeat,  // [B,32,128,128]
                              unsigned short* __restrict__ inp) {
    int bid = blockIdx.x;          // B * 16 * 16 blocks, 16x16 pixel tile
    int b   = bid >> 8;
    int rem = bid & 255;
    int y0  = (rem >> 4) << 4;
    int x0  = (rem & 15) << 4;
    int tid = threadIdx.x;
    int y = y0 + (tid >> 4);
    int x = x0 + (tid & 15);

    float refv[32];
    #pragma unroll
    for (int c = 0; c < 32; ++c)
        refv[c] = xr[(b * 32 + c) * HW + y * WW + x];

    unsigned short* op = inp + ((size_t)(b * PH + y + 1) * PW + x + 1) * 128;

    // --- FlowNet correlation, md=4, mean over C, then lrelu ---
    for (int dy = 0; dy < 9; ++dy) {
        int sy = y + dy - 4;
        bool yok = (unsigned)sy < (unsigned)HH;
        for (int dx = 0; dx < 9; ++dx) {
            int sx = x + dx - 4;
            float s = 0.0f;
            if (yok && (unsigned)sx < (unsigned)WW) {
                const float* np = xn + (size_t)b * 32 * HW + sy * WW + sx;
                #pragma unroll
                for (int c = 0; c < 32; ++c)
                    s = fmaf(refv[c], np[c * HW], s);
            }
            op[dy * 9 + dx] = f2bf(lrelu(s * (1.0f / 32.0f)));
        }
    }

    // --- bilinear 2x, align_corners=False (half-pixel centers), clamped ---
    int ky = y >> 1, kx = x >> 1;
    int iy0, iy1, ix0, ix1;
    float wy1, wx1;
    if (y & 1) { iy0 = ky; iy1 = (ky + 1 < 128) ? ky + 1 : 127; wy1 = 0.25f; }
    else       { iy0 = (ky - 1 >= 0) ? ky - 1 : 0; iy1 = ky;    wy1 = 0.75f; }
    if (x & 1) { ix0 = kx; ix1 = (kx + 1 < 128) ? kx + 1 : 127; wx1 = 0.25f; }
    else       { ix0 = (kx - 1 >= 0) ? kx - 1 : 0; ix1 = kx;    wx1 = 0.75f; }
    float wy0 = 1.0f - wy1, wx0 = 1.0f - wx1;

    #pragma unroll 4
    for (int c = 0; c < 32; ++c) {
        const float* fp = bfeat + (size_t)(b * 32 + c) * 128 * 128;
        float v = wy0 * (wx0 * fp[iy0 * 128 + ix0] + wx1 * fp[iy0 * 128 + ix1]) +
                  wy1 * (wx0 * fp[iy1 * 128 + ix0] + wx1 * fp[iy1 * 128 + ix1]);
        op[81 + c] = f2bf(v);
    }
    #pragma unroll
    for (int c = 0; c < 2; ++c) {
        const float* fp = bflow + (size_t)(b * 2 + c) * 128 * 128;
        float v = wy0 * (wx0 * fp[iy0 * 128 + ix0] + wx1 * fp[iy0 * 128 + ix1]) +
                  wy1 * (wx0 * fp[iy1 * 128 + ix0] + wx1 * fp[iy1 * 128 + ix1]);
        op[113 + c] = f2bf(v * 2.0f);
    }
    #pragma unroll
    for (int c = 115; c < 128; ++c) op[c] = 0;
}

// ---------------------------------------------------------------------------
// Implicit-GEMM 3x3 conv, WMMA bf16 -> f32, branch-free (halo-padded input),
// weights staged in LDS per block.
// One wave = 16-pixel strip (along W) x (NT*16) output channels.
// MODE 0: lrelu(acc+bias)                 -> bf16 padded NHWC   (conv0/conv1)
// MODE 1: lrelu(acc+bias+resid)           -> bf16 NHWC + f32 NCHW   (conv2)
// MODE 2: head (3 of 16 live): flows f32, sigmoid mask f32          (convf)
// ---------------------------------------------------------------------------
template <int CIN, int NT, int MODE>
__global__ void conv3x3_wmma_k(const unsigned short* __restrict__ in,
                               const unsigned short* __restrict__ wgt,
                               const float* __restrict__ bias,
                               unsigned short* __restrict__ outb,
                               const unsigned short* __restrict__ resid,
                               float* __restrict__ outfa,
                               float* __restrict__ outfb) {
    const int COUT = NT * 16;
    const int WELTS = 9 * COUT * CIN;          // bf16 elements of weight block
    __shared__ unsigned short smw[WELTS];      // <= 73728 B (gfx1250: 320KB WG)

    // cooperative weight stage: 16B per thread per step
    for (int i = threadIdx.x * 8; i < WELTS; i += blockDim.x * 8)
        *(uint4*)(smw + i) = *(const uint4*)(wgt + i);
    __syncthreads();

    int wave = blockIdx.x * (blockDim.x >> 5) + (threadIdx.x >> 5);
    int lane = threadIdx.x & 31;
    int m = lane & 15;      // A row / B,D column
    int g = lane >> 4;      // K half selector

    // strip decode: 16384 strips = B(4) * H(256) * (W/16 = 16)
    int xs = (wave & 15) << 4;
    int y  = (wave >> 4) & 255;
    int b  = wave >> 12;

    v8f acc[NT];
    #pragma unroll
    for (int nt = 0; nt < NT; ++nt)
        #pragma unroll
        for (int r = 0; r < 8; ++r) acc[nt][r] = 0.0f;

    #pragma unroll
    for (int ty = 0; ty < 3; ++ty) {
        // padded row: (y-1)+1 = y+ty ; padded col: (xs+m-1)+1 = xs+m+tx
        const unsigned short* rowb =
            in + ((size_t)(b * PH + y + ty) * PW + xs + m) * CIN;
        #pragma unroll
        for (int tx = 0; tx < 3; ++tx) {
            int t = ty * 3 + tx;
            const unsigned short* abase = rowb + (size_t)tx * CIN;
            #pragma unroll
            for (int kc = 0; kc < CIN; kc += 32) {
                v8bf lo = *(const v8bf*)(abase + kc + g * 8);
                v8bf hi = *(const v8bf*)(abase + kc + 16 + g * 8);
                v16bf afrag = __builtin_shufflevector(lo, hi,
                        0, 1, 2, 3, 4, 5, 6, 7, 8, 9, 10, 11, 12, 13, 14, 15);
                #pragma unroll
                for (int nt = 0; nt < NT; ++nt) {
                    const unsigned short* bp =
                        smw + (t * COUT + nt * 16 + m) * CIN + kc + g * 8;
                    v8bf blo = *(const v8bf*)bp;
                    v8bf bhi = *(const v8bf*)(bp + 16);
                    v16bf bfrag = __builtin_shufflevector(blo, bhi,
                            0, 1, 2, 3, 4, 5, 6, 7, 8, 9, 10, 11, 12, 13, 14, 15);     // ds_load_b128 x2
                    acc[nt] = __builtin_amdgcn_wmma_f32_16x16x32_bf16(
                        false, afrag, false, bfrag, (short)0, acc[nt], false, false);
                }
            }
        }
    }

    // Epilogue. D layout: lane group g, VGPR r -> pixel M = r + 8*g, N = nt*16+m.
    #pragma unroll
    for (int nt = 0; nt < NT; ++nt) {
        int n = nt * 16 + m;
        float bn = (MODE == 2) ? ((n < 3) ? bias[n] : 0.0f) : bias[n];
        #pragma unroll
        for (int r = 0; r < 8; ++r) {
            int px = xs + r + 8 * g;
            size_t P = (size_t)(b * PH + y + 1) * PW + px + 1;   // padded pixel
            float v = acc[nt][r] + bn;
            if (MODE == 0) {
                outb[P * COUT + n] = f2bf(lrelu(v));
            } else if (MODE == 1) {
                float tsum = v + bf2f(resid[P * 32 + n]);
                float f = lrelu(tsum);
                outb[P * 32 + n] = f2bf(f);
                outfa[(size_t)(b * 32 + n) * HW + y * WW + px] = f;  // feats NCHW
            } else {  // MODE == 2
                size_t pix = (size_t)y * WW + px;
                if (n == 0)      outfa[(size_t)(b * 2 + 0) * HW + pix] = v;
                else if (n == 1) outfa[(size_t)(b * 2 + 1) * HW + pix] = v;
                else if (n == 2) outfb[(size_t)b * HW + pix] = 1.0f / (1.0f + __expf(-v));
            }
        }
    }
}

// ---------------------------------------------------------------------------
// Launch
// ---------------------------------------------------------------------------
extern "C" void kernel_launch(void* const* d_in, const int* in_sizes, int n_in,
                              void* d_out, int out_size, void* d_ws, size_t ws_size,
                              hipStream_t stream) {
    const float* x_ref  = (const float*)d_in[0];
    const float* x_nb   = (const float*)d_in[1];
    const float* bflow  = (const float*)d_in[2];
    const float* bfeat  = (const float*)d_in[3];
    const float* w0 = (const float*)d_in[4];  const float* b0 = (const float*)d_in[5];
    const float* w1 = (const float*)d_in[6];  const float* b1 = (const float*)d_in[7];
    const float* w2 = (const float*)d_in[8];  const float* b2 = (const float*)d_in[9];
    const float* wf = (const float*)d_in[10]; const float* bf_ = (const float*)d_in[11];
    float* out = (float*)d_out;

    char* ws = (char*)d_ws;
    // Workspace layout (bytes), padded NHWC tensors:
    const size_t SZ_INP = (size_t)BB * PPX * 128 * 2;   // 68,161,536
    const size_t SZ_32  = (size_t)BB * PPX * 32 * 2;    // 17,040,384
    const size_t OFF_INP = 0;
    const size_t OFF_T1  = 0;                           // reuse inp region
    const size_t OFF_FB  = SZ_32;                       // still inside inp region
    const size_t OFF_H   = SZ_INP;
    const size_t OFF_W0  = OFF_H + SZ_32;
    const size_t OFF_W1  = OFF_W0 + 9 * 32 * 128 * 2;
    const size_t OFF_W2  = OFF_W1 + 9 * 32 * 32 * 2;
    const size_t OFF_WF  = OFF_W2 + 9 * 32 * 32 * 2;

    unsigned short* inp = (unsigned short*)(ws + OFF_INP);
    unsigned short* t1  = (unsigned short*)(ws + OFF_T1);
    unsigned short* fb  = (unsigned short*)(ws + OFF_FB);
    unsigned short* h   = (unsigned short*)(ws + OFF_H);
    unsigned short* w0r = (unsigned short*)(ws + OFF_W0);
    unsigned short* w1r = (unsigned short*)(ws + OFF_W1);
    unsigned short* w2r = (unsigned short*)(ws + OFF_W2);
    unsigned short* wfr = (unsigned short*)(ws + OFF_WF);

    // d_out layout: flows [B,2,1,H,W] | mask [B,1,H,W] | feats [B,32,H,W]
    float* flows = out;
    float* mask  = out + (size_t)BB * 2 * HW;
    float* feats = out + (size_t)BB * 3 * HW;

    const int hz_blocks = (BB * PPX + 255) / 256;

    prep_weights_k<<<64, 256, 0, stream>>>(w0, w1, w2, wf, w0r, w1r, w2r, wfr);
    zero_halo_k<<<hz_blocks, 256, 0, stream>>>(inp, 128);
    zero_halo_k<<<hz_blocks, 256, 0, stream>>>(h, 32);

    build_input_k<<<BB * 16 * 16, 256, 0, stream>>>(x_ref, x_nb, bflow, bfeat, inp);

    const int strips = BB * HH * (WW / 16);   // 16384
    const int blocks = strips / 8;            // 8 waves / 256-thread block

    // conv0: 128 -> 32, lrelu -> h
    conv3x3_wmma_k<128, 2, 0><<<blocks, 256, 0, stream>>>(
        inp, w0r, b0, h, nullptr, nullptr, nullptr);

    // t1/fb live in the (now dead) inp region: zero their halos after conv0.
    zero_halo_k<<<hz_blocks, 256, 0, stream>>>(t1, 32);
    zero_halo_k<<<hz_blocks, 256, 0, stream>>>(fb, 32);

    // conv1: 32 -> 32, lrelu -> t1
    conv3x3_wmma_k<32, 2, 0><<<blocks, 256, 0, stream>>>(
        h, w1r, b1, t1, nullptr, nullptr, nullptr);
    // conv2: 32 -> 32, +h residual, lrelu -> feats (f32 NCHW) + fb (bf16 NHWC)
    conv3x3_wmma_k<32, 2, 1><<<blocks, 256, 0, stream>>>(
        t1, w2r, b2, fb, h, feats, nullptr);
    // head: 32 -> 3 (padded to 16): flows + sigmoid mask
    conv3x3_wmma_k<32, 1, 2><<<blocks, 256, 0, stream>>>(
        fb, wfr, bf_, nullptr, nullptr, flows, mask);
}
